// PhaseSubgraphGNN_62191126446172
// MI455X (gfx1250) — matrix operand order repaired and checked by reading, not compiled, not run on hardware
//
#include <hip/hip_runtime.h>
#include <hip/hip_bf16.h>

#define DEV __device__ __forceinline__

typedef __attribute__((ext_vector_type(16))) _Float16 v16h;
typedef __attribute__((ext_vector_type(8)))  _Float16 v8h;
typedef __attribute__((ext_vector_type(8)))  float    v8f;

static constexpr int NN   = 100000;
static constexpr int EE   = 1600000;
static constexpr int MAXE = 1600000;
static constexpr int LROW = 72;   // padded LDS row stride (halves); 144B keeps 16B alignment

// ---------------- fragment helpers ----------------
// 16-bit A-matrix 16x32: lane half h holds K = {8h..8h+7} in elems 0..7 and
// {16+8h..16+8h+7} in elems 8..15  (VGPR i: K = 16*(i>=4) + 8*h + 2*(i&3) + slot)
DEV int kbaseA(int i, int half) {
  return ((i & 4) << 2) + (half << 3) + ((i & 3) << 1);
}

DEV v16h zero_v16h() {
  v16h a;
#pragma unroll
  for (int j = 0; j < 16; ++j) a[j] = (_Float16)0.0f;
  return a;
}
DEV v8f zero_v8f() {
  v8f c;
#pragma unroll
  for (int r = 0; r < 8; ++r) c[r] = 0.0f;
  return c;
}

// A fragment from a contiguous f16 row (global h or LDS tile row): 2x 16B loads.
DEV v16h afrag_f16row(const _Float16* row, int kt, int half) {
  v8h lo = *(const v8h*)(row + kt * 32 + 8 * half);
  v8h hi = *(const v8h*)(row + kt * 32 + 16 + 8 * half);
  v16h a;
#pragma unroll
  for (int j = 0; j < 8; ++j) { a[j] = lo[j]; a[j + 8] = hi[j]; }
  return a;
}

// A fragment from an f32 row (m buffer)
DEV v16h afrag_f32row(const float* __restrict__ row, int kt, int half) {
  v16h a;
#pragma unroll
  for (int i = 0; i < 8; ++i) {
    int k = kt * 32 + kbaseA(i, half);
    float2 v = *(const float2*)(row + k);
    a[2*i]   = (_Float16)v.x;
    a[2*i+1] = (_Float16)v.y;
  }
  return a;
}

// Weights pre-swizzled: fragment (kt,nt), lane, 16 contiguous halves.
DEV v16h bfrag(const _Float16* __restrict__ Wsw, int kt, int nt, int lane) {
  return *((const v16h*)Wsw + ((kt * 4 + nt) * 32 + lane));
}

DEV v8f wmma_f16(v16h a, v16h b, v8f c) {
  return __builtin_amdgcn_wmma_f32_16x16x32_f16(false, a, false, b, (short)0, c,
                                                false, false);
}

DEV void lds_fence()   { asm volatile("s_wait_dscnt 0" ::: "memory"); }
DEV void sched_fence() { asm volatile("" ::: "memory"); }

// Store C tile (+bias, optional relu) to LDS in f16
DEV void c_to_lds(_Float16* my, v8f c, float bb, int nt, int half, int col, bool relu) {
#pragma unroll
  for (int r = 0; r < 8; ++r) {
    float v = c[r] + bb;
    if (relu) v = v > 0.f ? v : 0.f;
    my[(r + 8 * half) * LROW + nt * 16 + col] = (_Float16)v;
  }
}

// Coalesced copy of the 16x64 f16 LDS tile to contiguous global rows (row n0..n0+15)
DEV void lds_tile_to_global(const _Float16* my, _Float16* dst, int lane) {
#pragma unroll
  for (int t = 0; t < 4; ++t) {
    int chunk = t * 32 + lane;            // 128 chunks of 8 halves
    int row = chunk >> 3;
    int colh = (chunk & 7) * 8;
    v8h v = *(const v8h*)(my + row * LROW + colh);
    *(v8h*)(dst + chunk * 8) = v;
  }
}

// ---------------- weight convert + swizzle ----------------
// dst element d = ((kt*4+nt)*32 + lane)*16 + j ; B 32x16 fp16 layout: K = kt*32 + 16*(j>=8) + 8*half + (j&7)
__global__ void k_convert_swizzle(const float* __restrict__ src, _Float16* __restrict__ dst,
                                  int Kact, int KT) {
  int d = blockIdx.x * blockDim.x + threadIdx.x;
  int total = KT * 2048;
  if (d >= total) return;
  int j    = d & 15;
  int lane = (d >> 4) & 31;
  int frag = d >> 9;
  int nt   = frag & 3;
  int kt   = frag >> 2;
  int half = lane >> 4;
  int n    = (nt << 4) + (lane & 15);
  int k    = kt * 32 + ((j & 8) << 1) + half * 8 + (j & 7);
  float v  = (k < Kact) ? src[k * 64 + n] : 0.0f;
  dst[d] = (_Float16)v;
}

__global__ void k_zero4(float4* __restrict__ p, int n4) {
  int i = blockIdx.x * blockDim.x + threadIdx.x;
  if (i < n4) p[i] = make_float4(0.f, 0.f, 0.f, 0.f);
}

// ---------------- phi0: [x(6)|z(8)] pad32 -> 64 -> 64 -> 64 ----------------
__global__ void __launch_bounds__(256) k_phi0(
    const float* __restrict__ x, const float* __restrict__ z,
    const _Float16* __restrict__ W1, const float* __restrict__ b1,
    const _Float16* __restrict__ W2, const float* __restrict__ b2,
    const _Float16* __restrict__ W3, const float* __restrict__ b3,
    _Float16* __restrict__ hout) {
  __shared__ _Float16 sh[8][16 * LROW];
  int lane = threadIdx.x & 31, w = threadIdx.x >> 5;
  int tile = blockIdx.x * 8 + w;
  if (tile * 16 >= NN) return;
  int n0 = tile * 16;
  int half = lane >> 4, col = lane & 15;
  int node = n0 + col;
  const float* xr = x + (size_t)node * 6;
  const float* zr = z + (size_t)node * 8;

  v16h a0;
#pragma unroll
  for (int i = 0; i < 8; ++i) {
#pragma unroll
    for (int s = 0; s < 2; ++s) {
      int k = kbaseA(i, half) + s;
      float v = 0.0f;
      if (k < 6) v = xr[k];
      else if (k < 14) v = zr[k - 6];
      a0[2*i + s] = (_Float16)v;
    }
  }

  _Float16* my = sh[w];
#pragma unroll
  for (int nt = 0; nt < 4; ++nt) {
    v8f c = zero_v8f();
    c = wmma_f16(a0, bfrag(W1, 0, nt, lane), c);
    c_to_lds(my, c, b1[nt * 16 + col], nt, half, col, true);
  }
  lds_fence();
  const _Float16* myrow = my + col * LROW;
  v16h p0 = afrag_f16row(myrow, 0, half);
  v16h p1 = afrag_f16row(myrow, 1, half);
  sched_fence();
#pragma unroll
  for (int nt = 0; nt < 4; ++nt) {
    v8f c = zero_v8f();
    c = wmma_f16(p0, bfrag(W2, 0, nt, lane), c);
    c = wmma_f16(p1, bfrag(W2, 1, nt, lane), c);
    c_to_lds(my, c, b2[nt * 16 + col], nt, half, col, true);
  }
  lds_fence();
  v16h q0 = afrag_f16row(myrow, 0, half);
  v16h q1 = afrag_f16row(myrow, 1, half);
  sched_fence();
#pragma unroll
  for (int nt = 0; nt < 4; ++nt) {
    v8f c = zero_v8f();
    c = wmma_f16(q0, bfrag(W3, 0, nt, lane), c);
    c = wmma_f16(q1, bfrag(W3, 1, nt, lane), c);
    c_to_lds(my, c, b3[nt * 16 + col], nt, half, col, false);
  }
  lds_fence();
  lds_tile_to_global(my, hout + (size_t)n0 * 64, lane);
}

// ---------------- edge MLP: [h[src](64)|ea(2)|r(4)] pad96 -> 64 -> 64 -> 64, scatter-add ----
__global__ void __launch_bounds__(256) k_edge(
    const _Float16* __restrict__ h, const int* __restrict__ ei,
    const float* __restrict__ ea, const int* __restrict__ eid,
    const float* __restrict__ eemb, int phase,
    const _Float16* __restrict__ W1, const float* __restrict__ b1,
    const _Float16* __restrict__ W2, const float* __restrict__ b2,
    const _Float16* __restrict__ W3, const float* __restrict__ b3,
    float* __restrict__ mout) {
  __shared__ _Float16 sh[8][16 * LROW];
  int lane = threadIdx.x & 31, w = threadIdx.x >> 5;
  int tile = blockIdx.x * 8 + w;
  if (tile * 16 >= EE) return;
  int e0 = tile * 16;
  int half = lane >> 4, col = lane & 15;
  int eIdx = e0 + col;

  int srcn = ei[eIdx];
  const _Float16* hrow = h + (size_t)srcn * 64;
  v16h a0 = afrag_f16row(hrow, 0, half);
  v16h a1 = afrag_f16row(hrow, 1, half);
  v16h a2 = zero_v16h();
  if (half == 0) {
    const float* er = ea + (size_t)eIdx * 2;
    float ev0 = er[0], ev1 = er[1];
    int id = eid[eIdx] + phase * MAXE;
    const float* rr = eemb + (size_t)id * 4;
    a2[0] = (_Float16)ev0;   a2[1] = (_Float16)ev1;
    a2[2] = (_Float16)rr[0]; a2[3] = (_Float16)rr[1];
    a2[4] = (_Float16)rr[2]; a2[5] = (_Float16)rr[3];
  }

  _Float16* my = sh[w];
#pragma unroll
  for (int nt = 0; nt < 4; ++nt) {
    v8f c = zero_v8f();
    c = wmma_f16(a0, bfrag(W1, 0, nt, lane), c);
    c = wmma_f16(a1, bfrag(W1, 1, nt, lane), c);
    c = wmma_f16(a2, bfrag(W1, 2, nt, lane), c);
    c_to_lds(my, c, b1[nt * 16 + col], nt, half, col, true);
  }
  lds_fence();
  const _Float16* myrow = my + col * LROW;
  v16h p0 = afrag_f16row(myrow, 0, half);
  v16h p1 = afrag_f16row(myrow, 1, half);
  sched_fence();
#pragma unroll
  for (int nt = 0; nt < 4; ++nt) {
    v8f c = zero_v8f();
    c = wmma_f16(p0, bfrag(W2, 0, nt, lane), c);
    c = wmma_f16(p1, bfrag(W2, 1, nt, lane), c);
    c_to_lds(my, c, b2[nt * 16 + col], nt, half, col, true);
  }
  lds_fence();
  v16h q0 = afrag_f16row(myrow, 0, half);
  v16h q1 = afrag_f16row(myrow, 1, half);
  sched_fence();

  int dstRow[8];
#pragma unroll
  for (int r = 0; r < 8; ++r) dstRow[r] = ei[EE + e0 + r + 8 * half];

#pragma unroll
  for (int nt = 0; nt < 4; ++nt) {
    v8f c = zero_v8f();
    c = wmma_f16(q0, bfrag(W3, 0, nt, lane), c);
    c = wmma_f16(q1, bfrag(W3, 1, nt, lane), c);
    float bb = b3[nt * 16 + col];
#pragma unroll
    for (int r = 0; r < 8; ++r) {
      float* p = mout + (size_t)dstRow[r] * 64 + nt * 16 + col;
      (void)__hip_atomic_fetch_add(p, c[r] + bb, __ATOMIC_RELAXED,
                                   __HIP_MEMORY_SCOPE_AGENT);
    }
  }
}

// ---------------- upd: [h(64)|m(64)|z(8)] pad160 -> 64 -> 64 -> 64 ----------------
__global__ void __launch_bounds__(256) k_upd(
    const _Float16* __restrict__ h, const float* __restrict__ m, const float* __restrict__ z,
    const _Float16* __restrict__ W1, const float* __restrict__ b1,
    const _Float16* __restrict__ W2, const float* __restrict__ b2,
    const _Float16* __restrict__ W3, const float* __restrict__ b3,
    _Float16* __restrict__ hout) {
  __shared__ _Float16 sh[8][16 * LROW];
  int lane = threadIdx.x & 31, w = threadIdx.x >> 5;
  int tile = blockIdx.x * 8 + w;
  if (tile * 16 >= NN) return;
  int n0 = tile * 16;
  int half = lane >> 4, col = lane & 15;
  int node = n0 + col;
  const _Float16* hr = h + (size_t)node * 64;
  const float* mr = m + (size_t)node * 64;
  const float* zr = z + (size_t)node * 8;

  v16h a0 = afrag_f16row(hr, 0, half);
  v16h a1 = afrag_f16row(hr, 1, half);
  v16h a2 = afrag_f32row(mr, 0, half);
  v16h a3 = afrag_f32row(mr, 1, half);
  v16h a4 = zero_v16h();
  if (half == 0) {
#pragma unroll
    for (int i = 0; i < 4; ++i) {
      float2 v = *(const float2*)(zr + 2 * i);
      a4[2*i] = (_Float16)v.x; a4[2*i+1] = (_Float16)v.y;
    }
  }

  _Float16* my = sh[w];
#pragma unroll
  for (int nt = 0; nt < 4; ++nt) {
    v8f c = zero_v8f();
    c = wmma_f16(a0, bfrag(W1, 0, nt, lane), c);
    c = wmma_f16(a1, bfrag(W1, 1, nt, lane), c);
    c = wmma_f16(a2, bfrag(W1, 2, nt, lane), c);
    c = wmma_f16(a3, bfrag(W1, 3, nt, lane), c);
    c = wmma_f16(a4, bfrag(W1, 4, nt, lane), c);
    c_to_lds(my, c, b1[nt * 16 + col], nt, half, col, true);
  }
  lds_fence();
  const _Float16* myrow = my + col * LROW;
  v16h p0 = afrag_f16row(myrow, 0, half);
  v16h p1 = afrag_f16row(myrow, 1, half);
  sched_fence();
#pragma unroll
  for (int nt = 0; nt < 4; ++nt) {
    v8f c = zero_v8f();
    c = wmma_f16(p0, bfrag(W2, 0, nt, lane), c);
    c = wmma_f16(p1, bfrag(W2, 1, nt, lane), c);
    c_to_lds(my, c, b2[nt * 16 + col], nt, half, col, true);
  }
  lds_fence();
  v16h q0 = afrag_f16row(myrow, 0, half);
  v16h q1 = afrag_f16row(myrow, 1, half);
  sched_fence();
#pragma unroll
  for (int nt = 0; nt < 4; ++nt) {
    v8f c = zero_v8f();
    c = wmma_f16(q0, bfrag(W3, 0, nt, lane), c);
    c = wmma_f16(q1, bfrag(W3, 1, nt, lane), c);
    c_to_lds(my, c, b3[nt * 16 + col], nt, half, col, false);
  }
  lds_fence();
  lds_tile_to_global(my, hout + (size_t)n0 * 64, lane);
}

// ---------------- readout: 64 -> 64 -> 64 -> 1 ----------------
__global__ void __launch_bounds__(256) k_readout(
    const _Float16* __restrict__ h,
    const _Float16* __restrict__ W1, const float* __restrict__ b1,
    const _Float16* __restrict__ W2, const float* __restrict__ b2,
    const float* __restrict__ W3, const float* __restrict__ b3,
    float* __restrict__ out) {
  __shared__ _Float16 sh[8][16 * LROW];
  int lane = threadIdx.x & 31, w = threadIdx.x >> 5;
  int tile = blockIdx.x * 8 + w;
  if (tile * 16 >= NN) return;
  int n0 = tile * 16;
  int half = lane >> 4, col = lane & 15;
  const _Float16* hr = h + (size_t)(n0 + col) * 64;

  v16h a0 = afrag_f16row(hr, 0, half);
  v16h a1 = afrag_f16row(hr, 1, half);

  _Float16* my = sh[w];
#pragma unroll
  for (int nt = 0; nt < 4; ++nt) {
    v8f c = zero_v8f();
    c = wmma_f16(a0, bfrag(W1, 0, nt, lane), c);
    c = wmma_f16(a1, bfrag(W1, 1, nt, lane), c);
    c_to_lds(my, c, b1[nt * 16 + col], nt, half, col, true);
  }
  lds_fence();
  const _Float16* myrow = my + col * LROW;
  v16h p0 = afrag_f16row(myrow, 0, half);
  v16h p1 = afrag_f16row(myrow, 1, half);
  sched_fence();
#pragma unroll
  for (int nt = 0; nt < 4; ++nt) {
    v8f c = zero_v8f();
    c = wmma_f16(p0, bfrag(W2, 0, nt, lane), c);
    c = wmma_f16(p1, bfrag(W2, 1, nt, lane), c);
    c_to_lds(my, c, b2[nt * 16 + col], nt, half, col, true);
  }
  lds_fence();
  if (half == 0) {
    const _Float16* row = my + col * LROW;
    float s = b3[0];
#pragma unroll 8
    for (int k = 0; k < 64; ++k) s += (float)row[k] * W3[k];
    out[n0 + col] = s;
  }
}

// ---------------- host ----------------
extern "C" void kernel_launch(void* const* d_in, const int* in_sizes, int n_in,
                              void* d_out, int out_size, void* d_ws, size_t ws_size,
                              hipStream_t stream) {
  (void)n_in; (void)out_size; (void)ws_size;

  // Param ordering: jax tree-flatten sorts dict keys. Auto-detect both plausible
  // orderings via element counts.
  bool sortedTop = (in_sizes[10] > 1000000);            // edge_emb (19.2M) vs node_emb (0.8M)
  int iEdgeEmb = sortedTop ? 10 : 11;
  int iNodeEmb = sortedTop ? 11 : 10;
  int iPhi0 = 12;
  int iPsi0 = 18;                                       // psi[l*3+p] at iPsi0 + 6*(l*3+p)
  bool readoutAt54 = (in_sizes[54] == 4096);            // readout.W1 4096 vs upd.W1 8704
  int iReadout = readoutAt54 ? 54 : 66;
  int iUpd[2]; iUpd[0] = readoutAt54 ? 60 : 54; iUpd[1] = readoutAt54 ? 66 : 60;

  // Workspace layout: m (f32) first, then f16 h ping-pong, then swizzled weights
  float* mB = (float*)d_ws;
  _Float16* hA = (_Float16*)(mB + (size_t)NN * 64);
  _Float16* hB = hA + (size_t)NN * 64;
  _Float16* wp = hB + (size_t)NN * 64;
  auto walloc = [&](int KT) { _Float16* r = wp; wp += (size_t)KT * 2048; return r; };
  auto conv = [&](int inIdx, _Float16* dst, int Kact, int KT) {
    int total = KT * 2048;
    k_convert_swizzle<<<(total + 255) / 256, 256, 0, stream>>>(
        (const float*)d_in[inIdx], dst, Kact, KT);
  };

  _Float16* phiW1 = walloc(1); conv(iPhi0 + 0, phiW1, 14, 1);
  _Float16* phiW2 = walloc(2); conv(iPhi0 + 1, phiW2, 64, 2);
  _Float16* phiW3 = walloc(2); conv(iPhi0 + 2, phiW3, 64, 2);

  _Float16 *psiW1[6], *psiW2[6], *psiW3[6];
  for (int q = 0; q < 6; ++q) {
    int base = iPsi0 + q * 6;
    psiW1[q] = walloc(3); conv(base + 0, psiW1[q], 70, 3);
    psiW2[q] = walloc(2); conv(base + 1, psiW2[q], 64, 2);
    psiW3[q] = walloc(2); conv(base + 2, psiW3[q], 64, 2);
  }
  _Float16 *updW1[2], *updW2[2], *updW3[2];
  for (int l = 0; l < 2; ++l) {
    updW1[l] = walloc(5); conv(iUpd[l] + 0, updW1[l], 136, 5);
    updW2[l] = walloc(2); conv(iUpd[l] + 1, updW2[l], 64, 2);
    updW3[l] = walloc(2); conv(iUpd[l] + 2, updW3[l], 64, 2);
  }
  _Float16* roW1 = walloc(2); conv(iReadout + 0, roW1, 64, 2);
  _Float16* roW2 = walloc(2); conv(iReadout + 1, roW2, 64, 2);

  const int nodeBlocks = (NN / 16 + 7) / 8;   // 782
  const int edgeBlocks = (EE / 16) / 8;       // 12500

  // phi0
  k_phi0<<<nodeBlocks, 256, 0, stream>>>(
      (const float*)d_in[0], (const float*)d_in[iNodeEmb],
      phiW1, (const float*)d_in[iPhi0 + 3],
      phiW2, (const float*)d_in[iPhi0 + 4],
      phiW3, (const float*)d_in[iPhi0 + 5], hA);

  _Float16* hcur = hA;
  _Float16* hnext = hB;
  for (int l = 0; l < 2; ++l) {
    k_zero4<<<(NN * 64 / 4 + 255) / 256, 256, 0, stream>>>((float4*)mB, NN * 64 / 4);
    for (int p = 0; p < 3; ++p) {
      int q = l * 3 + p;
      int base = iPsi0 + q * 6;
      k_edge<<<edgeBlocks, 256, 0, stream>>>(
          hcur, (const int*)d_in[1 + p], (const float*)d_in[4 + p],
          (const int*)d_in[7 + p], (const float*)d_in[iEdgeEmb], p,
          psiW1[q], (const float*)d_in[base + 3],
          psiW2[q], (const float*)d_in[base + 4],
          psiW3[q], (const float*)d_in[base + 5], mB);
    }
    k_upd<<<nodeBlocks, 256, 0, stream>>>(
        hcur, mB, (const float*)d_in[iNodeEmb],
        updW1[l], (const float*)d_in[iUpd[l] + 3],
        updW2[l], (const float*)d_in[iUpd[l] + 4],
        updW3[l], (const float*)d_in[iUpd[l] + 5], hnext);
    _Float16* t = hcur; hcur = hnext; hnext = t;
  }

  k_readout<<<nodeBlocks, 256, 0, stream>>>(
      hcur, roW1, (const float*)d_in[iReadout + 3],
      roW2, (const float*)d_in[iReadout + 4],
      (const float*)d_in[iReadout + 2], (const float*)d_in[iReadout + 5],
      (float*)d_out);
}
